// HashMapper_18691697672522
// MI455X (gfx1250) — 8-bit matrix operand order repaired and checked by reading, compiled, not run on hardware
//
#include <hip/hip_runtime.h>
#include <cstdint>

#define BATCH     32768
#define N_BITS    1024
#define HASH_BITS 14
#define N_HASH    3
#define TABLE     16384

typedef int v8i __attribute__((ext_vector_type(8)));

#define WS_WT     ((size_t)0)
#define WS_ADDR   ((size_t)(64 << 10))
#define WS_BITSU8 ((size_t)(1 << 20))
#define WS_PACK   ((size_t)(36 << 20))

__global__ void build_weights_kernel(const int* __restrict__ positions,
                                     unsigned char* __restrict__ wt) {
  int t = threadIdx.x;
  uint32_t* w32 = (uint32_t*)wt;
  for (int i = t; i < (16 * N_BITS) / 4; i += 256) w32[i] = 0u;
  __syncthreads();
  if (t < N_HASH * HASH_BITS) {
    int h = t / HASH_BITS;
    int j = t % HASH_BITS;
    int n = (N_BITS - 1) - positions[t];
    unsigned int weight = 1u << (HASH_BITS - 1 - j);
    if (weight < 256u)
      wt[(size_t)(2 * h) * N_BITS + n] = (unsigned char)weight;
    else
      wt[(size_t)(2 * h + 1) * N_BITS + n] = (unsigned char)(weight >> 8);
  }
}

__global__ void pack_bits_kernel(const int* __restrict__ bits,
                                 unsigned char* __restrict__ bitsu8) {
  size_t i4 = ((size_t)blockIdx.x * 256 + threadIdx.x) * 4;
  int4 v = *(const int4*)(bits + i4);
  unsigned int w = (unsigned int)(v.x & 1)
                 | ((unsigned int)(v.y & 1) << 8)
                 | ((unsigned int)(v.z & 1) << 16)
                 | ((unsigned int)(v.w & 1) << 24);
  *(unsigned int*)(bitsu8 + i4) = w;
}

__global__ void pack_tables_kernel(const float* __restrict__ tables,
                                   unsigned int* __restrict__ pack) {
  int gid  = blockIdx.x * 256 + threadIdx.x;
  int a    = gid & (TABLE - 1);
  int rest = gid >> 14;
  int w    = rest & 31;
  int h    = rest >> 5;
  const float* src = tables + (size_t)h * N_BITS * TABLE + a;
  unsigned int word = 0;
#pragma unroll
  for (int i = 0; i < 32; ++i) {
    float v = src[(size_t)(w * 32 + i) * TABLE];
    word |= (v != 0.0f ? 1u : 0u) << i;
  }
  pack[((size_t)h * TABLE + a) * 32 + w] = word;
}

__global__ void addr_wmma_kernel(const unsigned char* __restrict__ bitsu8,
                                 const unsigned char* __restrict__ wt,
                                 unsigned int* __restrict__ addrs) {
  int lane = threadIdx.x & 31;
  int wave = threadIdx.x >> 5;
  int row0 = (blockIdx.x * 8 + wave) * 16;
  int half = lane >> 4;
  int lm   = lane & 15;

  const unsigned char* aRow = bitsu8 + (size_t)(row0 + lm) * N_BITS + half * 8;
  const unsigned char* bRow = wt + (size_t)lm * N_BITS + half * 8;

  v8i acc = {};
  for (int kk = 0; kk < N_BITS; kk += 64) {
    const uint32_t* ap = (const uint32_t*)(aRow + kk);
    const uint32_t* bp = (const uint32_t*)(bRow + kk);
    v8i a, b;
    a[0] = ap[0];  a[1] = ap[1];  a[2] = ap[4];  a[3] = ap[5];
    a[4] = ap[8];  a[5] = ap[9];  a[6] = ap[12]; a[7] = ap[13];
    b[0] = bp[0];  b[1] = bp[1];  b[2] = bp[4];  b[3] = bp[5];
    b[4] = bp[8];  b[5] = bp[9];  b[6] = bp[12]; b[7] = bp[13];
    acc = __builtin_amdgcn_wmma_i32_16x16x64_iu8(false, a, false, b, acc,
                                                 false, false);
  }

#pragma unroll
  for (int r = 0; r < 8; ++r) {
    int v = acc[r];
    int partner = __shfl_xor(v, 1, 32);
    if ((lm & 1) == 0 && lm < 2 * N_HASH) {
      int h = lm >> 1;
      int m = r + half * 8;
      addrs[(size_t)(row0 + m) * 4 + h] = (unsigned int)(v + (partner << 8));
    }
  }
}

__global__ void vote_kernel(const unsigned int* __restrict__ pack,
                            const unsigned int* __restrict__ addrs,
                            float* __restrict__ out) {
  __shared__ unsigned int srows[3 * 32];
  int b = blockIdx.x;
  int t = threadIdx.x;
  if (t < 96) {
    int h = t >> 5;
    int w = t & 31;
    unsigned int a = addrs[(size_t)b * 4 + h];
    const unsigned int* row = pack + ((size_t)h * TABLE + a) * 32;
    __builtin_prefetch(row, 0, 0);
    srows[t] = row[w];
  }
  __syncthreads();
  unsigned int w0 = srows[t >> 3];
  unsigned int w1 = srows[32 + (t >> 3)];
  unsigned int w2 = srows[64 + (t >> 3)];
  unsigned int maj = (w0 & w1) | (w0 & w2) | (w1 & w2);
  int sh = (t & 7) * 4;
  float4 f;
  f.x = (float)((maj >> (sh + 0)) & 1u);
  f.y = (float)((maj >> (sh + 1)) & 1u);
  f.z = (float)((maj >> (sh + 2)) & 1u);
  f.w = (float)((maj >> (sh + 3)) & 1u);
  *(float4*)(out + (size_t)b * N_BITS + t * 4) = f;
}

extern "C" void kernel_launch(void* const* d_in, const int* in_sizes, int n_in,
                              void* d_out, int out_size, void* d_ws, size_t ws_size,
                              hipStream_t stream) {
  (void)in_sizes; (void)n_in; (void)out_size; (void)ws_size;

  const int*   bits      = (const int*)d_in[0];
  const float* tables    = (const float*)d_in[1];
  const int*   positions = (const int*)d_in[2];
  float*       out       = (float*)d_out;

  char* ws = (char*)d_ws;
  unsigned char* wt     = (unsigned char*)(ws + WS_WT);
  unsigned int*  addrs  = (unsigned int*)(ws + WS_ADDR);
  unsigned char* bitsu8 = (unsigned char*)(ws + WS_BITSU8);
  unsigned int*  pack   = (unsigned int*)(ws + WS_PACK);

  build_weights_kernel<<<1, 256, 0, stream>>>(positions, wt);

  pack_bits_kernel<<<(BATCH * N_BITS) / (256 * 4), 256, 0, stream>>>(bits, bitsu8);

  pack_tables_kernel<<<(N_HASH * 32 * TABLE) / 256, 256, 0, stream>>>(tables, pack);

  addr_wmma_kernel<<<(BATCH / 16) / 8, 256, 0, stream>>>(bitsu8, wt, addrs);

  vote_kernel<<<BATCH, 256, 0, stream>>>(pack, addrs, out);
}
